// SMART_77421080477940
// MI455X (gfx1250) — matrix operand order repaired and checked by reading, not compiled, not run on hardware
//
#include <hip/hip_runtime.h>

// ---------------------------------------------------------------------------
// SMART decoder block for MI455X (gfx1250), wave32 + WMMA bf16 + async LDS
// with double-buffered DMA pipeline.
//
// Prepass (memory-bound, one-time):
//   - tgt/feat0/feat1/feat2 converted fp32 -> bf16
//   - all weights converted + transposed to bf16 [N][K]
// Main pipeline (all GEMMs: v_wmma_f32_16x16x32_bf16, f32 accumulate; tiles
// DMA'd into LDS with global_load_async_to_lds_b128; next tile's DMA overlaps
// current tile's WMMAs via s_wait_asynccnt 0x4 + in-order async completion):
//   1. q   = tgt @ Wq + bq                       -> bf16   (16384 x 512)
//   2. kv0 = feat0 @ Wk0 + bk0                   -> bf16   (65536 x 1024)
//   3. kv1 = feat1 @ Wk1 + bk1                   -> bf16   (16384 x 1024)
//   4. kv2 = feat2 @ Wk2 + bk2                   -> bf16   ( 4096 x 1024)
//   5. xattn = sparse-6-key attention (gather)   -> bf16   (16384 x 512)
//   6. x1  = tgt + xattn @ Wp + bp               -> f32 + bf16 copies
//   7. h   = relu(x1 @ W1 + b1)                  -> bf16   (16384 x 2048)
//   8. out = x1 + h @ W2 + b2                    -> f32    (16384 x 512)
// ---------------------------------------------------------------------------

typedef __attribute__((ext_vector_type(16))) __bf16         v16bf;
typedef __attribute__((ext_vector_type(8)))  float          v8f;
typedef __attribute__((ext_vector_type(16))) unsigned short v16us;
typedef __attribute__((ext_vector_type(8)))  unsigned short v8us;
typedef __attribute__((ext_vector_type(4)))  unsigned short v4us;

static __device__ __forceinline__ unsigned short f2bf(float f) {
  union { float f; unsigned u; } v; v.f = f;
  unsigned r = v.u + 0x7FFFu + ((v.u >> 16) & 1u);   // round-to-nearest-even
  return (unsigned short)(r >> 16);
}
static __device__ __forceinline__ float bf2f(unsigned short u) {
  union { unsigned u; float f; } v; v.u = ((unsigned)u) << 16; return v.f;
}

// LDS aperture: flat addr[31:0] == wave-relative LDS byte offset (ISA 10.2).
static __device__ __forceinline__ unsigned lds_off(const void* p) {
  return (unsigned)(size_t)p;
}
// 16B/lane DMA into LDS, tracked by ASYNCcnt (gfx1250 async-copy path).
static __device__ __forceinline__ void async_copy_b128(const void* gsrc,
                                                       unsigned loff) {
  asm volatile("global_load_async_to_lds_b128 %0, %1, off"
               :: "v"(loff), "v"((unsigned long long)(size_t)gsrc)
               : "memory");
}
static __device__ __forceinline__ void wait_asynccnt0() {
  asm volatile("s_wait_asynccnt 0x0" ::: "memory");
}
// Async loads complete in order: cnt<=4 => the 4 older (current-tile) copies
// are done while the 4 newer (next-tile) copies may still be in flight.
static __device__ __forceinline__ void wait_asynccnt4() {
  asm volatile("s_wait_asynccnt 0x4" ::: "memory");
}

// ---------------------------------------------------------------------------
// Prepass: fp32 -> bf16 elementwise (vectorized x4)
// ---------------------------------------------------------------------------
__global__ __launch_bounds__(256) void cvt_bf16_kernel(
    const float* __restrict__ s, unsigned short* __restrict__ d, int n4)
{
  const int i = blockIdx.x * 256 + threadIdx.x;
  if (i < n4) {
    const float4 f = ((const float4*)s)[i];
    v4us p;
    p.x = f2bf(f.x); p.y = f2bf(f.y); p.z = f2bf(f.z); p.w = f2bf(f.w);
    ((v4us*)d)[i] = p;
  }
}

// ---------------------------------------------------------------------------
// Prepass: W[K][N] fp32 -> Wt[N][K] bf16 (32x32 LDS tiles)
// ---------------------------------------------------------------------------
__global__ __launch_bounds__(256) void transpose_bf16_kernel(
    const float* __restrict__ W, unsigned short* __restrict__ Wt, int K, int N)
{
  __shared__ __align__(16) unsigned short t[32][36];
  const int k0 = blockIdx.y * 32, n0 = blockIdx.x * 32;
  const int r  = threadIdx.x >> 3;
  const int c4 = (threadIdx.x & 7) * 4;
  const float4 f = *(const float4*)(W + (size_t)(k0 + r) * N + n0 + c4);
  t[c4 + 0][r] = f2bf(f.x);
  t[c4 + 1][r] = f2bf(f.y);
  t[c4 + 2][r] = f2bf(f.z);
  t[c4 + 3][r] = f2bf(f.w);
  __syncthreads();
  v4us o;
  o.x = t[r][c4 + 0]; o.y = t[r][c4 + 1]; o.z = t[r][c4 + 2]; o.w = t[r][c4 + 3];
  *(v4us*)(Wt + (size_t)(n0 + r) * K + k0 + c4) = o;
}

// ---------------------------------------------------------------------------
// GEMM:  C[M,N] = act(A[M,K] @ Wt[N,K]^T + bias[N]) (+ residual[M,N])
// A, Wt bf16 in global; 128x128 tile per 256-thread block (8 waves, 4x2);
// each wave: 32x64 sub-tile = 2x4 WMMA accumulators, K stepped by 32.
// Double-buffered LDS; tiles staged via global_load_async_to_lds_b128
// (4 async ops per wave per tile), next tile DMA overlapped with WMMAs.
// ---------------------------------------------------------------------------
#define TM 128
#define TN 128
#define TK 32
#define LDS_STRIDE 40                 // 32 + 8 pad ushorts -> 80B rows
#define TILE_US (TM * LDS_STRIDE)     // 5120 ushorts = 10240 B per tile buf

template <typename OutT, bool RELU, bool HAS_RES, bool HAS_AUX>
__global__ __launch_bounds__(256) void gemm_bf16_kernel(
    const unsigned short* __restrict__ A,    // [M][K] bf16
    const unsigned short* __restrict__ Wt,   // [N][K] bf16 (pre-transposed)
    const float* __restrict__ bias, const float* __restrict__ residual,
    OutT* __restrict__ C, unsigned short* __restrict__ C2,
    int M, int N, int K)
{
  __shared__ __align__(16) unsigned short As[2][TILE_US];
  __shared__ __align__(16) unsigned short Bs[2][TILE_US];

  const int tid  = threadIdx.x;
  const int lane = tid & 31;
  const int wid  = tid >> 5;
  const int wm   = wid & 3;       // wave row group: 4 x 32 rows
  const int wn   = wid >> 2;      // wave col group: 2 x 64 cols
  const int m0   = blockIdx.y * TM;
  const int n0   = blockIdx.x * TN;
  const int half = lane >> 4;
  const int r    = lane & 15;

  // per-thread staging coords: rows {srow, srow+64}, 16B sub-chunk `part`
  const int srow = tid >> 2;
  const int part = tid & 3;
  const unsigned short* ga0 = A  + (size_t)(m0 + srow) * K + part * 8;
  const unsigned short* gb0 = Wt + (size_t)(n0 + srow) * K + part * 8;
  const size_t rstep = (size_t)64 * K;
  const unsigned laoff = lds_off(&As[0][srow * LDS_STRIDE + part * 8]);
  const unsigned lboff = lds_off(&Bs[0][srow * LDS_STRIDE + part * 8]);
  const unsigned lrow64 = 64u * LDS_STRIDE * 2u;   // byte offset of row+64

  const v8f vzero = {};
  v8f acc[2][4];
#pragma unroll
  for (int i = 0; i < 2; ++i)
#pragma unroll
    for (int j = 0; j < 4; ++j) acc[i][j] = vzero;

  const int ktiles = K / TK;

  // prologue: DMA tile 0 into buffer 0
  async_copy_b128(ga0, laoff);
  async_copy_b128(ga0 + rstep, laoff + lrow64);
  async_copy_b128(gb0, lboff);
  async_copy_b128(gb0 + rstep, lboff + lrow64);

  for (int t = 0; t < ktiles; ++t) {
    const int cur = t & 1;
    if (t + 1 < ktiles) {
      // DMA next tile into the other buffer; overlaps this tile's WMMAs
      const unsigned boff = (cur ^ 1) ? (unsigned)(TILE_US * 2) : 0u;
      const int koff = (t + 1) * TK;
      async_copy_b128(ga0 + koff,         laoff + boff);
      async_copy_b128(ga0 + koff + rstep, laoff + boff + lrow64);
      async_copy_b128(gb0 + koff,         lboff + boff);
      async_copy_b128(gb0 + koff + rstep, lboff + boff + lrow64);
      wait_asynccnt4();   // current tile's 4 copies done (in-order)
    } else {
      wait_asynccnt0();
    }
    __syncthreads();

    // ---- fragments (CDNA5 16-bit WMMA lane layout), ds_load_b128 pairs ----
    v16bf afrag[2];
#pragma unroll
    for (int mi = 0; mi < 2; ++mi) {
      const unsigned short* p = &As[cur][(wm * 32 + mi * 16 + r) * LDS_STRIDE];
      v8us lo = *(const v8us*)(p + half * 8);        // K = half*8    .. +7
      v8us hi = *(const v8us*)(p + 16 + half * 8);   // K = 16+half*8 .. +7
      v16us tt = __builtin_shufflevector(lo, hi, 0, 1, 2, 3, 4, 5, 6, 7,
                                         8, 9, 10, 11, 12, 13, 14, 15);
      afrag[mi] = __builtin_bit_cast(v16bf, tt);
    }
    v16bf bfrag[4];
#pragma unroll
    for (int ni = 0; ni < 4; ++ni) {
      const unsigned short* p =
          &Bs[cur][(wn * 64 + ni * 16 + r) * LDS_STRIDE];
      v8us lo = *(const v8us*)(p + half * 8);        // K = half*8    .. +7
      v8us hi = *(const v8us*)(p + 16 + half * 8);   // K = 16+half*8 .. +7
      v16us tt = __builtin_shufflevector(lo, hi, 0, 1, 2, 3, 4, 5, 6, 7,
                                         8, 9, 10, 11, 12, 13, 14, 15);
      bfrag[ni] = __builtin_bit_cast(v16bf, tt);
    }
#pragma unroll
    for (int mi = 0; mi < 2; ++mi)
#pragma unroll
      for (int ni = 0; ni < 4; ++ni)
        acc[mi][ni] = __builtin_amdgcn_wmma_f32_16x16x32_bf16(
            false, afrag[mi], false, bfrag[ni], (short)0, acc[mi][ni],
            false, false);
    __syncthreads();   // reads retired before this buffer is DMA-overwritten
  }

  // ---- epilogue: bias (+relu) (+residual); f32/bf16 store (+bf16 aux) ----
  int   cols[4];
  float bcol[4];
#pragma unroll
  for (int ni = 0; ni < 4; ++ni) {
    cols[ni] = n0 + wn * 64 + ni * 16 + r;
    bcol[ni] = bias[cols[ni]];
  }
#pragma unroll
  for (int mi = 0; mi < 2; ++mi) {
#pragma unroll
    for (int v = 0; v < 8; ++v) {
      const int row = m0 + wm * 32 + mi * 16 + half * 8 + v;
      const size_t rb = (size_t)row * N;
#pragma unroll
      for (int ni = 0; ni < 4; ++ni) {
        float val = acc[mi][ni][v] + bcol[ni];
        if constexpr (RELU) val = val > 0.f ? val : 0.f;
        if constexpr (HAS_RES) val += residual[rb + cols[ni]];
        const size_t idx = rb + cols[ni];
        if constexpr (sizeof(OutT) == 4) C[idx] = val;
        else                             C[idx] = f2bf(val);
        if constexpr (HAS_AUX) C2[idx] = f2bf(val);
      }
    }
  }
}

// ---------------------------------------------------------------------------
// Sparse 6-key attention (reference has no softmax: x = sum_k (q.k) * v).
// One block per (b,l); one wave per head; each lane owns 2 of d=64 dims.
// ---------------------------------------------------------------------------
__global__ __launch_bounds__(256) void attn_kernel(
    const unsigned short* __restrict__ qb,
    const unsigned short* __restrict__ kv0,
    const unsigned short* __restrict__ kv1,
    const unsigned short* __restrict__ kv2,
    const int* __restrict__ attn_idx,
    unsigned short* __restrict__ xo)
{
  const int row  = blockIdx.x;            // b*4096 + l
  const int l    = row & 4095;
  const int b    = row >> 12;
  const int h    = threadIdx.x >> 5;      // head = wave id
  const int lane = threadIdx.x & 31;
  const int col  = h * 64 + lane * 2;

  const unsigned qraw = *(const unsigned*)&qb[(size_t)row * 512 + col];
  const float q0 = bf2f((unsigned short)(qraw & 0xFFFFu));
  const float q1 = bf2f((unsigned short)(qraw >> 16));

  float a0 = 0.f, a1 = 0.f;
#pragma unroll
  for (int kk = 0; kk < 6; ++kk) {
    const int t = attn_idx[l * 6 + kk];
    const unsigned short* base;
    size_t rkv;
    if (t < 16384)      { base = kv0; rkv = (size_t)b * 16384 + t; }
    else if (t < 20480) { base = kv1; rkv = (size_t)b * 4096 + (t - 16384); }
    else                { base = kv2; rkv = (size_t)b * 1024 + (t - 20480); }
    const unsigned kraw = *(const unsigned*)&base[rkv * 1024 + col];
    const unsigned vraw = *(const unsigned*)&base[rkv * 1024 + 512 + col];
    float dot = q0 * bf2f((unsigned short)(kraw & 0xFFFFu)) +
                q1 * bf2f((unsigned short)(kraw >> 16));
#pragma unroll
    for (int off = 16; off > 0; off >>= 1)
      dot += __shfl_xor(dot, off, 32);     // wave32 all-reduce
    a0 += dot * bf2f((unsigned short)(vraw & 0xFFFFu));
    a1 += dot * bf2f((unsigned short)(vraw >> 16));
  }
  const unsigned o = (unsigned)f2bf(a0) | ((unsigned)f2bf(a1) << 16);
  *(unsigned*)&xo[(size_t)row * 512 + col] = o;
}

// ---------------------------------------------------------------------------
// Launch
// ---------------------------------------------------------------------------
extern "C" void kernel_launch(void* const* d_in, const int* in_sizes, int n_in,
                              void* d_out, int out_size, void* d_ws, size_t ws_size,
                              hipStream_t stream) {
  (void)in_sizes; (void)n_in; (void)out_size; (void)ws_size;
  typedef unsigned short us;

  const float* tgt   = (const float*)d_in[0];
  const float* feat0 = (const float*)d_in[1];
  const float* feat1 = (const float*)d_in[2];
  const float* feat2 = (const float*)d_in[3];
  const int*   aidx  = (const int*)  d_in[4];
  const float* Wq  = (const float*)d_in[5];  const float* bq  = (const float*)d_in[6];
  const float* Wk0 = (const float*)d_in[7];  const float* bk0 = (const float*)d_in[8];
  const float* Wk1 = (const float*)d_in[9];  const float* bk1 = (const float*)d_in[10];
  const float* Wk2 = (const float*)d_in[11]; const float* bk2 = (const float*)d_in[12];
  const float* Wp  = (const float*)d_in[13]; const float* bp  = (const float*)d_in[14];
  const float* W1  = (const float*)d_in[15]; const float* b1  = (const float*)d_in[16];
  const float* W2  = (const float*)d_in[17]; const float* b2  = (const float*)d_in[18];
  float* out = (float*)d_out;

  // ---- workspace layout (bytes) ----
  char* ws = (char*)d_ws;
  us* tgtb  = (us*)(ws);                      // 16384x512  bf16   16,777,216
  us* f0b   = (us*)(ws +  16777216ull);       // 65536x256  bf16   33,554,432
  us* f1b   = (us*)(ws +  50331648ull);       // 16384x512  bf16   16,777,216
  us* f2b   = (us*)(ws +  67108864ull);       //  4096x1024 bf16    8,388,608
  us* WqT   = (us*)(ws +  75497472ull);       //   512x512          524,288
  us* Wk0T  = (us*)(ws +  76021760ull);       //  1024x256          524,288
  us* Wk1T  = (us*)(ws +  76546048ull);       //  1024x512        1,048,576
  us* Wk2T  = (us*)(ws +  77594624ull);       //  1024x1024       2,097,152
  us* WpT   = (us*)(ws +  79691776ull);       //   512x512          524,288
  us* W1T   = (us*)(ws +  80216064ull);       //  2048x512        2,097,152
  us* W2T   = (us*)(ws +  82313216ull);       //   512x2048       2,097,152
  us* q_b   = (us*)(ws +  84410368ull);       // 16384x512  bf16  16,777,216
  us* kv0   = (us*)(ws + 101187584ull);       // 65536x1024 bf16 134,217,728
  us* kv1   = (us*)(ws + 235405312ull);       // 16384x1024 bf16  33,554,432
  us* kv2   = (us*)(ws + 268959744ull);       //  4096x1024 bf16   8,388,608
  us* xatb  = (us*)(ws + 277348352ull);       // 16384x512  bf16  16,777,216
  float* x1 = (float*)(ws + 294125568ull);    // 16384x512  f32   33,554,432
  us* x1b   = (us*)(ws + 327680000ull);       // 16384x512  bf16  16,777,216
  us* h_b   = (us*)(ws + 344457216ull);       // 16384x2048 bf16  67,108,864
  // total: 411,566,080 bytes

  const dim3 blk(256);

  // ---- prepass: activations fp32 -> bf16 ----
  cvt_bf16_kernel<<<dim3(8388608 / 4 / 256),  blk, 0, stream>>>(tgt,   tgtb, 8388608 / 4);
  cvt_bf16_kernel<<<dim3(16777216 / 4 / 256), blk, 0, stream>>>(feat0, f0b, 16777216 / 4);
  cvt_bf16_kernel<<<dim3(8388608 / 4 / 256),  blk, 0, stream>>>(feat1, f1b,  8388608 / 4);
  cvt_bf16_kernel<<<dim3(4194304 / 4 / 256),  blk, 0, stream>>>(feat2, f2b,  4194304 / 4);
  // ---- prepass: weights fp32 [K][N] -> bf16 [N][K] ----
  transpose_bf16_kernel<<<dim3(512 / 32,  512 / 32),  blk, 0, stream>>>(Wq,  WqT,  512, 512);
  transpose_bf16_kernel<<<dim3(1024 / 32, 256 / 32),  blk, 0, stream>>>(Wk0, Wk0T, 256, 1024);
  transpose_bf16_kernel<<<dim3(1024 / 32, 512 / 32),  blk, 0, stream>>>(Wk1, Wk1T, 512, 1024);
  transpose_bf16_kernel<<<dim3(1024 / 32, 1024 / 32), blk, 0, stream>>>(Wk2, Wk2T, 1024, 1024);
  transpose_bf16_kernel<<<dim3(512 / 32,  512 / 32),  blk, 0, stream>>>(Wp,  WpT,  512, 512);
  transpose_bf16_kernel<<<dim3(2048 / 32, 512 / 32),  blk, 0, stream>>>(W1,  W1T,  512, 2048);
  transpose_bf16_kernel<<<dim3(512 / 32,  2048 / 32), blk, 0, stream>>>(W2,  W2T,  2048, 512);

  // ---- 1) q = tgt @ Wq + bq ----
  gemm_bf16_kernel<us, false, false, false>
      <<<dim3(512 / TN, 16384 / TM), blk, 0, stream>>>(
          tgtb, WqT, bq, nullptr, q_b, nullptr, 16384, 512, 512);
  // ---- 2-4) kv projections ----
  gemm_bf16_kernel<us, false, false, false>
      <<<dim3(1024 / TN, 65536 / TM), blk, 0, stream>>>(
          f0b, Wk0T, bk0, nullptr, kv0, nullptr, 65536, 1024, 256);
  gemm_bf16_kernel<us, false, false, false>
      <<<dim3(1024 / TN, 16384 / TM), blk, 0, stream>>>(
          f1b, Wk1T, bk1, nullptr, kv1, nullptr, 16384, 1024, 512);
  gemm_bf16_kernel<us, false, false, false>
      <<<dim3(1024 / TN, 4096 / TM), blk, 0, stream>>>(
          f2b, Wk2T, bk2, nullptr, kv2, nullptr, 4096, 1024, 1024);
  // ---- 5) gathered 6-key attention ----
  attn_kernel<<<dim3(16384), blk, 0, stream>>>(q_b, kv0, kv1, kv2, aidx, xatb);
  // ---- 6) x1 = tgt + xattn @ Wp + bp  (f32 + bf16 aux) ----
  gemm_bf16_kernel<float, false, true, true>
      <<<dim3(512 / TN, 16384 / TM), blk, 0, stream>>>(
          xatb, WpT, bp, tgt, x1, x1b, 16384, 512, 512);
  // ---- 7) h = relu(x1 @ W1 + b1) ----
  gemm_bf16_kernel<us, true, false, false>
      <<<dim3(2048 / TN, 16384 / TM), blk, 0, stream>>>(
          x1b, W1T, b1, nullptr, h_b, nullptr, 16384, 2048, 512);
  // ---- 8) out = x1 + h @ W2 + b2 ----
  gemm_bf16_kernel<float, false, true, false>
      <<<dim3(512 / TN, 16384 / TM), blk, 0, stream>>>(
          h_b, W2T, b2, x1, out, nullptr, 16384, 512, 2048);
}
